// WindowAttention_14929306321540
// MI455X (gfx1250) — matrix operand order repaired and checked by reading, not compiled, not run on hardware
//
#include <hip/hip_runtime.h>
#include <hip/hip_bf16.h>

// ---- problem constants (match reference) ----
#define DIMC   128
#define HEADS  4
#define HD     32
#define NTOK   49
#define NPAD   64
#define NWIN   64
#define NB     4096
#define LDX    136    // f16 row stride for token-major LDS buffers (64 rows)
#define LDV    72     // f16 row stride for vT (128 rows x 64 tokens)
#define LDATT  68     // f32 row stride for attention scores (4*64 rows)

typedef __attribute__((ext_vector_type(16))) _Float16 v16h;
typedef __attribute__((ext_vector_type(8)))  _Float16 v8h;
typedef __attribute__((ext_vector_type(8)))  float    v8f;
typedef __attribute__((ext_vector_type(4)))  float    v4f;

#define WMMA_F16(a,b,c) __builtin_amdgcn_wmma_f32_16x16x32_f16(false,(a),false,(b),(short)0,(c),false,false)

// A-matrix fragment (16x32 f16, M x K). ISA 7.12.2: lanes 0-15 M=0-15 (g=0),
// lanes 16-31 M=0-15 (g=1). VGPR0-3 hold K = g*8+0..7, VGPR4-7 hold K = 16+g*8+0..7.
// base points at tile origin (row0*ld + k0). Two 16B loads per lane.
__device__ __forceinline__ v16h frag_a(const _Float16* base, int ld, int lane) {
  int m = lane & 15, g = lane >> 4;
  const _Float16* r = base + m * ld + g * 8;
  v8h lo = *(const v8h*)(r);
  v8h hi = *(const v8h*)(r + 16);
  v16h a;
#pragma unroll
  for (int t = 0; t < 8; ++t) { a[t] = lo[t]; a[t + 8] = hi[t]; }
  return a;
}

// B-matrix fragment (32x16 f16, K x N) from a B-transposed ([n][k] row-major) buffer.
// Lanes 0-15: N=lane, K=0-7,16-23; lanes 16-31: N=lane-16, K=8-15,24-31.
// Lane reads halves at row n, cols g*8 and 16+g*8: two 16B loads.
__device__ __forceinline__ v16h frag_bt(const _Float16* base, int ld, int lane) {
  int n = lane & 15, g = lane >> 4;
  const _Float16* r = base + n * ld + g * 8;
  v8h lo = *(const v8h*)(r);
  v8h hi = *(const v8h*)(r + 16);
  v16h b;
#pragma unroll
  for (int t = 0; t < 8; ++t) { b[t] = lo[t]; b[t + 8] = hi[t]; }
  return b;
}

// ---------------- prep: weight transpose+f16 convert, rel-bias gather ----------------
__global__ void prep_kernel(const float* __restrict__ qkv_w,      // [128,384]
                            const float* __restrict__ proj_w,     // [128,128]
                            const float* __restrict__ bias_table, // [169,4]
                            const int*   __restrict__ rel_idx,    // [49,49]
                            _Float16* __restrict__ wqkvT,  // [384][128]
                            _Float16* __restrict__ wprojT, // [128][128]
                            float* __restrict__ relb)      // [4][49][49]
{
  int idx = blockIdx.x * 256 + threadIdx.x;
  if (idx < 384 * 128) {
    int n = idx >> 7, k = idx & 127;
    wqkvT[idx] = (_Float16)qkv_w[k * 384 + n];
  } else if (idx < 384 * 128 + 128 * 128) {
    int t = idx - 384 * 128;
    int n = t >> 7, k = t & 127;
    wprojT[t] = (_Float16)proj_w[k * 128 + n];
  } else if (idx < 384 * 128 + 128 * 128 + HEADS * NTOK * NTOK) {
    int t = idx - (384 * 128 + 128 * 128);
    int h = t / (NTOK * NTOK), r = t % (NTOK * NTOK);
    relb[t] = bias_table[rel_idx[r] * HEADS + h];
  }
}

// ---------------- fused per-window attention ----------------
// LDS layout (halves):
//   xA  [64][136]  f16  @ 0        (aliased by os after QKV stage)
//   qs  [64][136]  f16  @ 8704     (Q * scale, token-major)
//   ks  [64][136]  f16  @ 17408    (K token-major == B^T layout for Q.K^T)
//   vT  [128][72]  f16  @ 26112    (V transposed == B^T layout for P.V)
//   att [4][64][68] f32 @ byte 70656 (scores; softmaxed P rewritten in place as f16)
#define SMEM_BYTES 140288

__global__ __launch_bounds__(256)
void win_attn_kernel(const float* __restrict__ x,      // [4096,49,128]
                     const float* __restrict__ mask,   // [64,49,49]
                     const float* __restrict__ qkv_b,  // [384]
                     const float* __restrict__ proj_b, // [128]
                     const _Float16* __restrict__ wqkvT,  // [384][128]
                     const _Float16* __restrict__ wprojT, // [128][128]
                     const float* __restrict__ relb,      // [4][49][49]
                     float* __restrict__ out)             // [4096,49,128]
{
  extern __shared__ char smem[];
  _Float16* xA  = (_Float16*)smem;
  _Float16* osb = xA;                 // alias, safe: xA dead after QKV stage
  _Float16* qs  = xA + 64 * LDX;
  _Float16* ks  = qs + 64 * LDX;
  _Float16* vT  = ks + 64 * LDX;
  float*    att = (float*)(smem + 2 * (3 * 64 * LDX + 128 * LDV));

  const int tid  = threadIdx.x;
  // wave index is uniform per wave: force it into an SGPR so all tile indices /
  // base addresses are scalar and store-path branches compile to s_cbranch.
  const int wave = __builtin_amdgcn_readfirstlane(tid >> 5);
  const int lane = tid & 31;
  const int n    = lane & 15;         // C-fragment column within tile
  const int g    = lane >> 4;         // C-fragment row-half selector
  const int b    = blockIdx.x;
  const int wid  = b & (NWIN - 1);
  const float SCALE = 0.17677669529663687f;   // 32^-0.5

  // ---- stage 1: x -> LDS f16 (8 halves / store), rows 49..63 zeroed ----
  {
    const float* xb = x + (size_t)b * NTOK * DIMC;
    for (int c = tid; c < 64 * 16; c += 256) {   // 8 floats per chunk
      int row = c >> 4, c8 = (c & 15) * 8;
      v8h hv;
      if (row < NTOK) {
        v4f f0 = *(const v4f*)(xb + row * DIMC + c8);
        v4f f1 = *(const v4f*)(xb + row * DIMC + c8 + 4);
#pragma unroll
        for (int t = 0; t < 4; ++t) { hv[t] = (_Float16)f0[t]; hv[t + 4] = (_Float16)f1[t]; }
      } else {
#pragma unroll
        for (int t = 0; t < 8; ++t) hv[t] = (_Float16)0.f;
      }
      *(v8h*)(xA + row * LDX + c8) = hv;
    }
  }
  __syncthreads();

  // ---- stage 2: QKV GEMM  [64x128] @ [128x384] + bias ----
  // waves own 3 output-column tiles; B fragments (from L2) reused across 4 M tiles.
#pragma unroll
  for (int q = 0; q < 3; ++q) {
    const int nt  = wave * 3 + q;                         // SGPR
    const int seg = nt >> 3;                              // 0=Q 1=K 2=V (uniform)
    v16h bf[4];
#pragma unroll
    for (int kk = 0; kk < 4; ++kk)
      bf[kk] = frag_bt(wqkvT + nt * 16 * DIMC + kk * 32, DIMC, lane);
    const int col = nt * 16 + n;
    const float bias = qkv_b[col];
#pragma unroll
    for (int mt = 0; mt < 4; ++mt) {
      v8f c = {};
#pragma unroll
      for (int kk = 0; kk < 4; ++kk) {
        v16h a = frag_a(xA + mt * 16 * LDX + kk * 32, LDX, lane);
        c = WMMA_F16(a, bf[kk], c);
      }
      const int row0 = mt * 16 + 8 * g;
      if (seg == 0) {                          // Q: scale applied here
#pragma unroll
        for (int r = 0; r < 8; ++r)
          qs[(row0 + r) * LDX + col] = (_Float16)((c[r] + bias) * SCALE);
      } else if (seg == 1) {                   // K token-major
#pragma unroll
        for (int r = 0; r < 8; ++r)
          ks[(row0 + r) * LDX + (col - 128)] = (_Float16)(c[r] + bias);
      } else {                                 // V transposed -> contiguous 16B store
        v8h t;
#pragma unroll
        for (int r = 0; r < 8; ++r) t[r] = (_Float16)(c[r] + bias);
        *(v8h*)(vT + (col - 256) * LDV + row0) = t;
      }
    }
  }
  __syncthreads();

  // ---- stage 3: scores = (Q*scale) . K^T per head  (K-dim = 32 -> 1 WMMA/tile) ----
#pragma unroll
  for (int j = 0; j < 8; ++j) {
    const int t = wave * 8 + j;               // 64 tiles: 4 heads x 4x4 (all SGPR)
    const int h = t >> 4, mi = (t >> 2) & 3, nj = t & 3;
    v16h a  = frag_a (qs + mi * 16 * LDX + h * 32, LDX, lane);
    v16h bb = frag_bt(ks + nj * 16 * LDX + h * 32, LDX, lane);
    v8f c = {};
    c = WMMA_F16(a, bb, c);
#pragma unroll
    for (int r = 0; r < 8; ++r)
      att[(h * 64 + mi * 16 + 8 * g + r) * LDATT + nj * 16 + n] = c[r];
  }
  __syncthreads();

  // ---- stage 4: softmax (+rel bias +mask); write P back in place as f16 ----
  {
    const int h = tid >> 6, i = tid & 63;
    float* row = att + (h * 64 + i) * LDATT;
    _Float16* prow = (_Float16*)row;          // f16 view of same row
    if (i < NTOK) {
      const float* mrow = mask + (size_t)wid * NTOK * NTOK + i * NTOK;
      const float* rrow = relb + h * NTOK * NTOK + i * NTOK;
      float mx = -1e30f;
      for (int j = 0; j < NTOK; ++j) {
        float s = row[j] + rrow[j] + mrow[j];
        row[j] = s;
        mx = fmaxf(mx, s);
      }
      float sum = 0.f;
      for (int j = 0; j < NTOK; ++j) {
        float e = __expf(row[j] - mx);
        row[j] = e;
        sum += e;
      }
      const float inv = 1.f / sum;
      for (int j = 0; j < 64; ++j)            // write at 2j < read at 4j: safe in-place
        prow[j] = (j < NTOK) ? (_Float16)(row[j] * inv) : (_Float16)0.f;
    } else {
      for (int j = 0; j < 64; ++j) prow[j] = (_Float16)0.f;  // padded rows: P = 0
    }
  }
  __syncthreads();

  // ---- stage 5: out = P . V per head  (K-dim = 64 -> 2 WMMAs/tile) ----
  {
    const _Float16* P = (const _Float16*)att; // f16 view, row stride 136 halves
#pragma unroll
    for (int j = 0; j < 4; ++j) {
      const int t = wave * 4 + j;             // 32 tiles: 4 heads x 4x2
      const int h = t >> 3, mi = (t >> 1) & 3, nd = t & 1;
      v8f c = {};
#pragma unroll
      for (int kk = 0; kk < 2; ++kk) {
        v16h a  = frag_a (P  + (h * 64 + mi * 16) * LDX + kk * 32, LDX, lane);
        v16h bb = frag_bt(vT + (h * 32 + nd * 16) * LDV + kk * 32, LDV, lane);
        c = WMMA_F16(a, bb, c);
      }
#pragma unroll
      for (int r = 0; r < 8; ++r)
        osb[(mi * 16 + 8 * g + r) * LDX + h * 32 + nd * 16 + n] = (_Float16)c[r];
    }
  }
  __syncthreads();

  // ---- stage 6: projection  [64x128] @ [128x128] + bias -> global ----
  {
    float* ob = out + (size_t)b * NTOK * DIMC;
#pragma unroll
    for (int j = 0; j < 4; ++j) {
      const int t = wave * 4 + j;             // 32 tiles: 4x8
      const int mt = t >> 3, nt = t & 7;
      v8f c = {};
#pragma unroll
      for (int kk = 0; kk < 4; ++kk) {
        v16h a  = frag_a (osb + mt * 16 * LDX + kk * 32, LDX, lane);
        v16h bb = frag_bt(wprojT + nt * 16 * DIMC + kk * 32, DIMC, lane);
        c = WMMA_F16(a, bb, c);
      }
      const float bias = proj_b[nt * 16 + n];
      const int row0 = mt * 16 + 8 * g;
#pragma unroll
      for (int r = 0; r < 8; ++r) {
        const int tok = row0 + r;
        if (tok < NTOK) ob[tok * DIMC + nt * 16 + n] = c[r] + bias;
      }
    }
  }
}

extern "C" void kernel_launch(void* const* d_in, const int* in_sizes, int n_in,
                              void* d_out, int out_size, void* d_ws, size_t ws_size,
                              hipStream_t stream) {
  const float* x          = (const float*)d_in[0];
  const float* mask       = (const float*)d_in[1];
  const float* qkv_w      = (const float*)d_in[2];
  const float* qkv_b      = (const float*)d_in[3];
  const float* proj_w     = (const float*)d_in[4];
  const float* proj_b     = (const float*)d_in[5];
  const float* bias_table = (const float*)d_in[6];
  const int*   rel_idx    = (const int*)d_in[7];
  float* out = (float*)d_out;

  // workspace: wqkvT f16 [384][128] | wprojT f16 [128][128] | relb f32 [4][49][49]
  _Float16* wqkvT  = (_Float16*)d_ws;
  _Float16* wprojT = wqkvT + 384 * 128;
  float*    relb   = (float*)((char*)d_ws + 2 * (384 * 128 + 128 * 128));

  const int prep_items = 384 * 128 + 128 * 128 + HEADS * NTOK * NTOK;
  prep_kernel<<<(prep_items + 255) / 256, 256, 0, stream>>>(
      qkv_w, proj_w, bias_table, rel_idx, wqkvT, wprojT, relb);

  win_attn_kernel<<<NB, 256, SMEM_BYTES, stream>>>(
      x, mask, qkv_b, proj_b, wqkvT, wprojT, relb, out);
}
